// kNN_16810501997049
// MI455X (gfx1250) — matrix operand order repaired and checked by reading, not compiled, not run on hardware
//
#include <hip/hip_runtime.h>

// ---------------------------------------------------------------------------
// 1-NN classifier for MI455X (gfx1250, wave32, WMMA).
//   score[i,j] = ||y_j||^2 - 2 * <x_i, y_j>   (x^2 / sqrt / clamp are monotone
//   or constant per row -> same argmin as the reference's cdist+argmin)
//   prediction[i] = labels[argmin_j score[i,j]]   (k=1 vote collapses)
//
// GEMM runs on v_wmma_f32_16x16x32_bf16 with 3-term compensated bf16 split
// (hi*hi + hi*lo + lo*hi) for ~f32 accuracy at bf16 matrix-pipe speed.
// ---------------------------------------------------------------------------

typedef __attribute__((ext_vector_type(16))) __bf16 v16bf;
typedef __attribute__((ext_vector_type(4)))  __bf16 v4bf;
typedef __attribute__((ext_vector_type(8)))  float  v8f;

#define N_IN   4096
#define M_DATA 32768
#define DDIM   256

// -------- kernel 0a: split data f32 -> bf16 hi/lo planes (one pass) --------
__global__ __launch_bounds__(256) void cvt_kernel(const float* __restrict__ src,
                                                  __bf16* __restrict__ hi,
                                                  __bf16* __restrict__ lo,
                                                  int n4) {
  int i = blockIdx.x * blockDim.x + threadIdx.x;
  if (i >= n4) return;
  const float4 v = ((const float4*)src)[i];
  float x[4] = {v.x, v.y, v.z, v.w};
  v4bf h, l;
#pragma unroll
  for (int e = 0; e < 4; ++e) {
    __bf16 hh = (__bf16)x[e];
    h[e] = hh;
    l[e] = (__bf16)(x[e] - (float)hh);
  }
  *(v4bf*)(hi + (size_t)i * 4) = h;   // 8B aligned packed stores
  *(v4bf*)(lo + (size_t)i * 4) = l;
}

// -------- kernel 0b: exact f32 row norms of data (one wave per row) --------
__global__ __launch_bounds__(256) void y2_kernel(const float* __restrict__ data,
                                                 float* __restrict__ y2) {
  const int wave = threadIdx.x >> 5, lane = threadIdx.x & 31;
  const int row  = blockIdx.x * 8 + wave;
  const float4* p = (const float4*)(data + (size_t)row * DDIM);
  float4 a = p[lane * 2], b = p[lane * 2 + 1];
  float s = a.x*a.x + a.y*a.y + a.z*a.z + a.w*a.w
          + b.x*b.x + b.y*b.y + b.z*b.z + b.w*b.w;
#pragma unroll
  for (int m = 16; m >= 1; m >>= 1) s += __shfl_xor(s, m, 32);
  if (lane == 0) y2[row] = s;
}

// -------- main kernel: fused WMMA GEMM + running argmin + label pick -------
// Grid: 128 WGs x 256 threads (8 waves). WG owns 32 input rows.
// Wave (rt,ct): rt = wave&1 row tile (16 rows), ct = wave>>1 col tile;
// sweeps data columns in steps of 64 (4 col tiles x 16).
__global__ __launch_bounds__(256) void knn_kernel(const float*  __restrict__ input,
                                                  const __bf16* __restrict__ dhi,
                                                  const __bf16* __restrict__ dlo,
                                                  const float*  __restrict__ y2,
                                                  const float*  __restrict__ labels,
                                                  float*        __restrict__ out) {
  __shared__ float sV[32][4];
  __shared__ int   sI[32][4];

  const int tid  = threadIdx.x;
  const int wave = tid >> 5;
  const int lane = tid & 31;
  const int half = lane >> 4;   // K-half selector in 16-bit WMMA layouts
  const int l15  = lane & 15;
  const int rt   = wave & 1;
  const int ct   = wave >> 1;
  const int rowBase = blockIdx.x * 32 + rt * 16;

  // A fragments: 16 rows x 256 K as bf16 hi/lo, 8 chunks of K=32, cached in
  // VGPRs. A order used here: lane l15 = row M; per half-wave the 16 packed
  // elements are K 0-15 / 16-31 of each 32-chunk (contiguous, matches B).
  v16bf ahi[8], alo[8];
  {
    const float* arow = input + (size_t)(rowBase + l15) * DDIM;
#pragma unroll
    for (int kc = 0; kc < 8; ++kc) {
#pragma unroll
      for (int g = 0; g < 2; ++g) {
        const int kb = kc * 32 + half * 16 + g * 8;
#pragma unroll
        for (int e = 0; e < 8; ++e) {
          float x  = arow[kb + e];
          __bf16 h = (__bf16)x;
          ahi[kc][g * 8 + e] = h;
          alo[kc][g * 8 + e] = (__bf16)(x - (float)h);
        }
      }
    }
  }

  float bestv[8];
  int   besti[8];
#pragma unroll
  for (int r = 0; r < 8; ++r) { bestv[r] = 3.4e38f; besti[r] = 0x7fffffff; }

  for (int mb = 0; mb < M_DATA; mb += 64) {
    const int jc = mb + ct * 16 + l15;     // data row == C column of this lane
    // B order used here: lane l15 = column N, half selects K 0-15 / 16-31 of
    // each 32-chunk -> 16 contiguous bf16 (32B, aligned) per plane per chunk.
    const __bf16* bh = dhi + (size_t)jc * DDIM + half * 16;
    const __bf16* bl = dlo + (size_t)jc * DDIM + half * 16;

    if (mb + 64 < M_DATA) {                // uniform branch, EXEC stays full
      __builtin_prefetch(bh + 64 * DDIM, 0, 1);   // global_prefetch_b8
      __builtin_prefetch(bl + 64 * DDIM, 0, 1);
    }

    v8f c0 = {}, c1 = {}, c2 = {};         // 3 accumulators: ILP across WMMAs
#pragma unroll
    for (int kc = 0; kc < 8; ++kc) {
      v16bf bhi = *(const v16bf*)(bh + kc * 32);
      v16bf blo = *(const v16bf*)(bl + kc * 32);
      c0 = __builtin_amdgcn_wmma_f32_16x16x32_bf16(false, ahi[kc], false, bhi,
                                                   (short)0, c0, false, false);
      c1 = __builtin_amdgcn_wmma_f32_16x16x32_bf16(false, ahi[kc], false, blo,
                                                   (short)0, c1, false, false);
      c2 = __builtin_amdgcn_wmma_f32_16x16x32_bf16(false, alo[kc], false, bhi,
                                                   (short)0, c2, false, false);
    }

    const float yy = y2[jc];
#pragma unroll
    for (int r = 0; r < 8; ++r) {          // C layout: VGPR r = row r + 8*half
      float s = yy - 2.0f * (c0[r] + c1[r] + c2[r]);
      if (s < bestv[r] || (s == bestv[r] && jc < besti[r])) {
        bestv[r] = s; besti[r] = jc;
      }
    }
  }

  // Reduce argmin across the 16 columns of each half-wave (masks<16 keep the
  // two halves, which hold different rows, separate).
#pragma unroll
  for (int r = 0; r < 8; ++r) {
    float v = bestv[r]; int idx = besti[r];
#pragma unroll
    for (int m = 1; m < 16; m <<= 1) {
      float ov = __shfl_xor(v, m, 32);
      int   oi = __shfl_xor(idx, m, 32);
      if (ov < v || (ov == v && oi < idx)) { v = ov; idx = oi; }
    }
    if (l15 == 0) {
      sV[rt * 16 + half * 8 + r][ct] = v;
      sI[rt * 16 + half * 8 + r][ct] = idx;
    }
  }
  __syncthreads();

  // Reduce across the 4 column-tile waves; emit label of nearest neighbor.
  if (tid < 32) {
    float v = sV[tid][0]; int idx = sI[tid][0];
#pragma unroll
    for (int q = 1; q < 4; ++q) {
      float ov = sV[tid][q]; int oi = sI[tid][q];
      if (ov < v || (ov == v && oi < idx)) { v = ov; idx = oi; }
    }
    out[blockIdx.x * 32 + tid] = labels[idx];
  }
  if (blockIdx.x == 0 && tid == 0) out[N_IN] = 0.0f;  // second (scalar) output
}

// ---------------------------------------------------------------------------
extern "C" void kernel_launch(void* const* d_in, const int* in_sizes, int n_in,
                              void* d_out, int out_size, void* d_ws, size_t ws_size,
                              hipStream_t stream) {
  (void)in_sizes; (void)n_in; (void)out_size; (void)ws_size;
  const float* input  = (const float*)d_in[0];   // [4096,256]
  const float* data   = (const float*)d_in[1];   // [32768,256]
  const float* labels = (const float*)d_in[2];   // [32768]
  float* out = (float*)d_out;                    // 4096 + 1 floats

  // workspace: dhi (16MB) | dlo (16MB) | y2 (128KB)  => ~33.7MB
  const size_t planeBytes = (size_t)M_DATA * DDIM * sizeof(__bf16);
  __bf16* dhi = (__bf16*)d_ws;
  __bf16* dlo = (__bf16*)((char*)d_ws + planeBytes);
  float*  y2  = (float*)((char*)d_ws + 2 * planeBytes);

  const int n4 = M_DATA * DDIM / 4;              // 2,097,152 float4s
  cvt_kernel<<<n4 / 256, 256, 0, stream>>>(data, dhi, dlo, n4);
  y2_kernel<<<M_DATA / 8, 256, 0, stream>>>(data, y2);
  knn_kernel<<<N_IN / 32, 256, 0, stream>>>(input, dhi, dlo, y2, labels, out);
}
